// PathologyGraph_19937238188708
// MI455X (gfx1250) — compile-verified, matrix operand order checked by reading, not run on hardware
//
#include <hip/hip_runtime.h>

// ---------------------------------------------------------------------------
// MI455X (gfx1250, wave32) implementation.
// Dominant cost: 151 MB HBM traffic (~6.5us @23.3TB/s). Batch GEMM is folded
// from [8192,768]x[768,768] into [8192,768]x[768,48] because q is only used
// against batch-independent k.  Scores GEMM uses v_wmma_f32_16x16x32_bf16.
// ---------------------------------------------------------------------------

typedef __attribute__((ext_vector_type(16))) __bf16 v16bf;
typedef __attribute__((ext_vector_type(8)))  float  v8f;

#define D      768
#define NCLS   5
#define NH     8
#define HD     96
#define BROWS  8192
#define NCOLP  48      // NH*NCLS = 40 padded to 48 (3 WMMA N-tiles)
#define KTILES 24      // 768 / 32
#define NTILES 3
#define RPB    64      // rows per block (4 waves x 16 rows)

// ---------------- prep kernels (all tiny, batch-independent) ----------------

// edge_feat[i,d] = (1/3) * sum_r (sum_j A[i,j,r]*node_emb[j,d]) * edge_emb[r,d]
// Adjacency is a fixed constant -> hardcoded neighbor sums.
__global__ void k_edgefeat(const float* __restrict__ ne,
                           const float* __restrict__ ee,
                           float* __restrict__ ef) {
  int t = blockIdx.x * blockDim.x + threadIdx.x;
  if (t >= NCLS * D) return;
  int i = t / D, d = t % D;
  float e0 = ee[0 * D + d], e1 = ee[1 * D + d], e2 = ee[2 * D + d];
  float n0 = ne[0 * D + d], n1 = ne[1 * D + d], n2 = ne[2 * D + d], n3 = ne[3 * D + d];
  float s0 = 0.f, s1 = 0.f;
  switch (i) {
    case 0: s0 = n1 + n2; break;
    case 1: s0 = n0 + n2; break;
    case 2: s0 = n0 + n1 + n3; break;
    case 3: s0 = n2; s1 = n2; break;  // hierarchical edge 3->2 (rel 1)
    default: break;
  }
  float s2 = ne[i * D + d];  // self loop (rel 2)
  ef[t] = (s0 * e0 + s1 * e1 + s2 * e2) * (1.0f / 3.0f);
}

// h = [node_emb, edge_feat] @ w1 + b1   (w1 stored [in=2D, out=D] row-major)
__global__ void k_h(const float* __restrict__ ne, const float* __restrict__ ef,
                    const float* __restrict__ w1, const float* __restrict__ b1,
                    float* __restrict__ h) {
  int t = blockIdx.x * blockDim.x + threadIdx.x;
  if (t >= NCLS * D) return;
  int i = t / D, o = t % D;
  float acc = b1[o];
  for (int c = 0; c < D; ++c) acc += ne[i * D + c] * w1[c * D + o];
  for (int c = 0; c < D; ++c) acc += ef[i * D + c] * w1[(D + c) * D + o];
  h[t] = acc;
}

// LayerNorm over last dim + ReLU. One block per node row.
__global__ void k_ln(const float* __restrict__ h, const float* __restrict__ g,
                     const float* __restrict__ b, float* __restrict__ hn) {
  __shared__ float r1[256], r2[256];
  int i = blockIdx.x;
  float s = 0.f, s2 = 0.f;
  for (int c = threadIdx.x; c < D; c += blockDim.x) {
    float v = h[i * D + c];
    s += v; s2 += v * v;
  }
  r1[threadIdx.x] = s; r2[threadIdx.x] = s2;
  __syncthreads();
  for (int off = 128; off > 0; off >>= 1) {
    if (threadIdx.x < off) {
      r1[threadIdx.x] += r1[threadIdx.x + off];
      r2[threadIdx.x] += r2[threadIdx.x + off];
    }
    __syncthreads();
  }
  float mu  = r1[0] * (1.0f / D);
  float var = r2[0] * (1.0f / D) - mu * mu;
  float inv = rsqrtf(var + 1e-5f);
  for (int c = threadIdx.x; c < D; c += blockDim.x) {
    float v = (h[i * D + c] - mu) * inv * g[c] + b[c];
    hn[i * D + c] = v > 0.f ? v : 0.f;
  }
}

// out[5,768] = A[5,768] @ W[768,768] + bias   (W stored [in,out])
__global__ void k_mm5(const float* __restrict__ A, const float* __restrict__ W,
                      const float* __restrict__ bias, float* __restrict__ out) {
  int t = blockIdx.x * blockDim.x + threadIdx.x;
  if (t >= NCLS * D) return;
  int i = t / D, o = t % D;
  float acc = bias[o];
  for (int c = 0; c < D; ++c) acc += A[i * D + c] * W[c * D + o];
  out[t] = acc;
}

// Fold k into wq:  W2[c, j=(h*5+n)] = (1/sqrt(96)) * sum_d wq[c,96h+d]*k[n,96h+d]
// sbias[j]        = (1/sqrt(96)) * sum_d bq[96h+d]*k[n,96h+d]
// W2 is written directly in the v_wmma_f32_16x16x32_bf16 B-operand lane layout:
//   lane = half*16 + n_in_tile;  elems 0..7 <-> K = 8*half + e,
//                                elems 8..15 <-> K = 16 + 8*half + (e-8)
__global__ void k_buildB(const float* __restrict__ wq, const float* __restrict__ bq,
                         const float* __restrict__ km, __bf16* __restrict__ wsB,
                         float* __restrict__ sbias) {
  int t = blockIdx.x * blockDim.x + threadIdx.x;
  if (t >= D * NCOLP) return;
  int c = t / NCOLP, j = t % NCOLP;
  const float scale = 0.1020620726f;  // 1/sqrt(96)
  float val = 0.f;
  if (j < NH * NCLS) {
    int hh = j / NCLS, n = j % NCLS;
    float s = 0.f;
    for (int d = 0; d < HD; ++d)
      s += wq[c * D + hh * HD + d] * km[n * D + hh * HD + d];
    val = s * scale;
    if (c == 0) {
      float sb = 0.f;
      for (int d = 0; d < HD; ++d)
        sb += bq[hh * HD + d] * km[n * D + hh * HD + d];
      sbias[j] = sb * scale;
    }
  } else if (c == 0) {
    sbias[j] = 0.f;  // padding columns
  }
  int ktile = c >> 5, kin = c & 31;
  int ntile = j >> 4, nin = j & 15;
  int hlf, e;
  if      (kin <  8) { hlf = 0; e = kin;            }
  else if (kin < 16) { hlf = 1; e = kin - 8;        }
  else if (kin < 24) { hlf = 0; e = 8 + (kin - 16); }
  else               { hlf = 1; e = 8 + (kin - 24); }
  int lane = hlf * 16 + nin;
  wsB[(((ktile * NTILES + ntile) * 32) + lane) * 16 + e] = (__bf16)val;
}

// ---------------- main fused kernel ----------------
// Per wave: 16 batch rows. scores[16,48] = bf16-WMMA( x_tile , W2 ) + sbias,
// then per-row head-softmax -> head-mean -> presence-softmax -> act,
// then block-cooperative float4 store of activated = updated * act.
__global__ __launch_bounds__(128) void k_main(
    const float* __restrict__ x, const float* __restrict__ cp,
    const float* __restrict__ upd, const float* __restrict__ sbias,
    const __bf16* __restrict__ wsB,
    float* __restrict__ outAct, float* __restrict__ outA) {
  __shared__ float s_upd[NCLS * D];           // 15 KB: updated[5,768]
  __shared__ float s_stage[4][16][NCOLP];     // 12 KB: per-wave score tiles
  __shared__ float s_act[RPB][8];             //  2 KB: act per row

  const int tid  = threadIdx.x;
  const int wave = tid >> 5;
  const int lane = tid & 31;

  for (int i = tid; i < NCLS * D; i += blockDim.x) s_upd[i] = upd[i];

  const int rowBase = blockIdx.x * RPB;
  const int row0    = rowBase + wave * 16;
  const int m       = lane & 15;       // A-matrix row within tile
  const int hlf     = lane >> 4;       // K-half select
  const int kbase   = hlf * 8;
  const float* xrow = x + (size_t)(row0 + m) * D;

  v8f acc0 = {}, acc1 = {}, acc2 = {};
  const v16bf* Bbase = (const v16bf*)wsB;

  for (int kt = 0; kt < KTILES; ++kt) {
    // A fragment: ISA 16-bit A layout (16x32). elems 0..7: K=kbase+e,
    // elems 8..15: K=16+kbase+(e-8).  Two 16B-aligned float4 pairs -> bf16.
    const float* xp = xrow + kt * 32 + kbase;
    float4 x0 = *(const float4*)(xp + 0);
    float4 x1 = *(const float4*)(xp + 4);
    float4 x2 = *(const float4*)(xp + 16);
    float4 x3 = *(const float4*)(xp + 20);
    v16bf a;
    a[0] = (__bf16)x0.x; a[1] = (__bf16)x0.y; a[2]  = (__bf16)x0.z; a[3]  = (__bf16)x0.w;
    a[4] = (__bf16)x1.x; a[5] = (__bf16)x1.y; a[6]  = (__bf16)x1.z; a[7]  = (__bf16)x1.w;
    a[8] = (__bf16)x2.x; a[9] = (__bf16)x2.y; a[10] = (__bf16)x2.z; a[11] = (__bf16)x2.w;
    a[12] = (__bf16)x3.x; a[13] = (__bf16)x3.y; a[14] = (__bf16)x3.z; a[15] = (__bf16)x3.w;

    const v16bf* bp = Bbase + (size_t)(kt * NTILES) * 32 + lane;
    v16bf b0 = bp[0], b1 = bp[32], b2 = bp[64];
    acc0 = __builtin_amdgcn_wmma_f32_16x16x32_bf16(false, a, false, b0, (short)0, acc0, false, false);
    acc1 = __builtin_amdgcn_wmma_f32_16x16x32_bf16(false, a, false, b1, (short)0, acc1, false, false);
    acc2 = __builtin_amdgcn_wmma_f32_16x16x32_bf16(false, a, false, b2, (short)0, acc2, false, false);
  }

  // C/D layout: VGPR r, lanes 0-15 -> M=r, N=lane; lanes 16-31 -> M=r+8.
  const int ncol = lane & 15;
  const float sb0 = sbias[ncol], sb1 = sbias[16 + ncol], sb2 = sbias[32 + ncol];
#pragma unroll
  for (int r = 0; r < 8; ++r) {
    int mr = r + 8 * hlf;
    s_stage[wave][mr][ncol]      = acc0[r] + sb0;
    s_stage[wave][mr][16 + ncol] = acc1[r] + sb1;
    s_stage[wave][mr][32 + ncol] = acc2[r] + sb2;
  }
  __syncthreads();

  if (lane < 16) {
    const int row = row0 + lane;
    const float* s = &s_stage[wave][lane][0];
    float aw[NCLS] = {0.f, 0.f, 0.f, 0.f, 0.f};
#pragma unroll
    for (int hh = 0; hh < NH; ++hh) {
      float v[NCLS], mx = -3.4e38f;
#pragma unroll
      for (int n = 0; n < NCLS; ++n) { v[n] = s[hh * NCLS + n]; mx = fmaxf(mx, v[n]); }
      float e[NCLS], sum = 0.f;
#pragma unroll
      for (int n = 0; n < NCLS; ++n) { e[n] = __expf(v[n] - mx); sum += e[n]; }
      float inv = 1.0f / sum;
#pragma unroll
      for (int n = 0; n < NCLS; ++n) aw[n] += e[n] * inv;
    }
    const float* cpr = cp + (size_t)row * NCLS;
    float p[NCLS], mx = -3.4e38f;
#pragma unroll
    for (int n = 0; n < NCLS; ++n) {
      p[n] = (aw[n] * 0.125f) * (cpr[n] + 0.1f);
      mx = fmaxf(mx, p[n]);
    }
    float e[NCLS], sum = 0.f;
#pragma unroll
    for (int n = 0; n < NCLS; ++n) { e[n] = __expf(p[n] - mx); sum += e[n]; }
    float inv = 1.0f / sum;
    float* ao = outA + (size_t)row * NCLS;
    int lr = wave * 16 + lane;
#pragma unroll
    for (int n = 0; n < NCLS; ++n) {
      float a = e[n] * inv;
      ao[n] = a;            // second output: act [B,5]
      s_act[lr][n] = a;
    }
  }
  __syncthreads();

  // activated[b,n,:] = updated[n,:] * act[b,n] -- coalesced float4 stores.
  const int total4 = RPB * NCLS * (D / 4);   // 61440 float4 per block
  const float4* u4 = (const float4*)s_upd;
  for (int idx = tid; idx < total4; idx += blockDim.x) {
    int d4 = idx % (D / 4);
    int t2 = idx / (D / 4);
    int n  = t2 % NCLS;
    int r  = t2 / NCLS;
    float a  = s_act[r][n];
    float4 u = u4[n * (D / 4) + d4];
    float4 o = make_float4(u.x * a, u.y * a, u.z * a, u.w * a);
    ((float4*)outAct)[((size_t)(rowBase + r) * NCLS + n) * (D / 4) + d4] = o;
  }
}

// ---------------- launch ----------------
extern "C" void kernel_launch(void* const* d_in, const int* in_sizes, int n_in,
                              void* d_out, int out_size, void* d_ws, size_t ws_size,
                              hipStream_t stream) {
  (void)in_sizes; (void)n_in; (void)out_size; (void)ws_size;
  const float* x        = (const float*)d_in[0];
  const float* cp       = (const float*)d_in[1];
  const float* node_emb = (const float*)d_in[2];
  const float* edge_emb = (const float*)d_in[3];
  const float* w1       = (const float*)d_in[4];
  const float* b1       = (const float*)d_in[5];
  const float* ln_g     = (const float*)d_in[6];
  const float* ln_b     = (const float*)d_in[7];
  const float* w2       = (const float*)d_in[8];
  const float* b2       = (const float*)d_in[9];
  const float* wq       = (const float*)d_in[10];
  const float* wk       = (const float*)d_in[11];
  const float* bq       = (const float*)d_in[13];
  const float* bk       = (const float*)d_in[14];
  // wv/bv/wo/bo (d_in[12,15,16,17]) are dead: attended output is discarded.

  float* out    = (float*)d_out;
  float* outAct = out;                               // activated [8192,5,768]
  float* outA   = out + (size_t)BROWS * NCLS * D;    // act       [8192,5]

  char* w = (char*)d_ws;
  float* ef  = (float*)w; w += NCLS * D * sizeof(float);
  float* h   = (float*)w; w += NCLS * D * sizeof(float);
  float* hn  = (float*)w; w += NCLS * D * sizeof(float);
  float* upd = (float*)w; w += NCLS * D * sizeof(float);
  float* km  = (float*)w; w += NCLS * D * sizeof(float);
  float* sb  = (float*)w; w += 64 * sizeof(float);        // 48 used, 32B-align pad
  __bf16* wsB = (__bf16*)w;                               // 768*48 bf16, WMMA-B layout

  k_edgefeat<<<30, 128, 0, stream>>>(node_emb, edge_emb, ef);
  k_h       <<<30, 128, 0, stream>>>(node_emb, ef, w1, b1, h);
  k_ln      <<<NCLS, 256, 0, stream>>>(h, ln_g, ln_b, hn);
  k_mm5     <<<30, 128, 0, stream>>>(hn, w2, b2, upd);    // updated
  k_mm5     <<<30, 128, 0, stream>>>(upd, wk, bk, km);    // k
  k_buildB  <<<144, 256, 0, stream>>>(wq, bq, km, wsB, sb);
  k_main    <<<BROWS / RPB, 128, 0, stream>>>(x, cp, upd, sb, wsB, outAct, outA);
}
